// StarSimplePairWiseClassifier_59382217835114
// MI455X (gfx1250) — matrix operand order repaired and checked by reading, not compiled, not run on hardware
//
#include <hip/hip_runtime.h>
#include <hip/hip_bf16.h>

#define Bz 8
#define NSP 128
#define NC 24
#define NNB 5
#define Dd 1536
#define Hh 128
#define NP 276                      // 24*23/2
#define NROWS (Bz*NC + Bz*NC*NNB)   // 192 + 960 = 1152 unique feature rows
#define NCROWS (Bz*NP)              // 2208 center-pair rows
#define NNROWS (Bz*NP*25)           // 55200 neighbor-pair rows
#define MBIG (NCROWS + NNROWS)      // 57408 total pair rows (divisible by 16)

typedef __attribute__((ext_vector_type(16))) _Float16 v16h;
typedef __attribute__((ext_vector_type(8)))  _Float16 v8h;
typedef __attribute__((ext_vector_type(8)))  float    v8f;

union V16 { v16h v; v8h h[2]; };

// ---------------------------------------------------------------------------
// fragment helpers (ISA 16-bit A/B layouts for v_wmma_f32_16x16x32_f16)
// ---------------------------------------------------------------------------
__device__ __forceinline__ void load_a(V16& a, const _Float16* arow, int k0, int ka)
{
    a.h[0] = *(const v8h*)(arow + k0 + ka);
    a.h[1] = *(const v8h*)(arow + k0 + ka + 16);
}

__device__ __forceinline__ void load_pa(V16& a, const _Float16* fr, const _Float16* sr,
                                        int k0, int ka)
{
    a.h[0] = (*(const v8h*)(fr + k0 + ka))      * (*(const v8h*)(sr + k0 + ka));
    a.h[1] = (*(const v8h*)(fr + k0 + ka + 16)) * (*(const v8h*)(sr + k0 + ka + 16));
}

// 4 B fragments (n-tiles g..g+3): compile-time K -> base + immediate offsets
template<int K>
__device__ __forceinline__ void load_b4(V16* b, const _Float16* bk, int kb, int g)
{
    #pragma unroll
    for (int n = 0; n < 4; ++n) {
        b[n].h[0] = *(const v8h*)(bk + (g + n)*16*K + kb);
        b[n].h[1] = *(const v8h*)(bk + (g + n)*16*K + kb + 16);
    }
}

__device__ __forceinline__ void wmma4(v8f* acc, const V16& a, const V16* b)
{
    #pragma unroll
    for (int n = 0; n < 4; ++n)
        acc[n] = __builtin_amdgcn_wmma_f32_16x16x32_f16(false, a.v, false, b[n].v,
                                                        (short)0, acc[n], false, false);
}

// ---------------------------------------------------------------------------
// 1) c = center_map @ x ; neighbors = neighbor_map @ x   (one block per (b,ctr))
// ---------------------------------------------------------------------------
__global__ void agg_kernel(const float* __restrict__ x,
                           const float* __restrict__ cmap,
                           const float* __restrict__ nmap,
                           float* __restrict__ cOut,      // [B*24][D]
                           float* __restrict__ nbOut)     // [B*24*5][D]
{
    int bc = blockIdx.x;                 // b*24 + ctr
    int b  = bc / NC;
    const float* xb = x + (size_t)b * NSP * Dd;
    const float* cm = cmap + (size_t)bc * NSP;
    const float* nm = nmap + (size_t)bc * NNB * NSP;
    int t = threadIdx.x;                 // 256 threads * 6 = 1536
    float accC[6] = {0.f,0.f,0.f,0.f,0.f,0.f};
    float accN[NNB][6] = {};
    for (int s = 0; s < NSP; ++s) {
        float wc = cm[s];
        float w0 = nm[0*NSP + s], w1 = nm[1*NSP + s], w2 = nm[2*NSP + s];
        float w3 = nm[3*NSP + s], w4 = nm[4*NSP + s];
        const float* xr = xb + (size_t)s * Dd + t * 6;
        #pragma unroll
        for (int j = 0; j < 6; ++j) {
            float xv = xr[j];
            accC[j]    += wc * xv;
            accN[0][j] += w0 * xv;  accN[1][j] += w1 * xv;
            accN[2][j] += w2 * xv;  accN[3][j] += w3 * xv;
            accN[4][j] += w4 * xv;
        }
    }
    #pragma unroll
    for (int j = 0; j < 6; ++j) cOut[(size_t)bc * Dd + t*6 + j] = accC[j];
    for (int n = 0; n < NNB; ++n)
        #pragma unroll
        for (int j = 0; j < 6; ++j)
            nbOut[((size_t)bc * NNB + n) * Dd + t*6 + j] = accN[n][j];
}

// neighbor_mask = neighbor_map.sum(-1)
__global__ void mask_kernel(const float* __restrict__ nmap, float* __restrict__ nmask)
{
    int i = blockIdx.x * blockDim.x + threadIdx.x;
    if (i >= Bz*NC*NNB) return;
    const float* p = nmap + (size_t)i * NSP;
    float s = 0.f;
    for (int k = 0; k < NSP; ++k) s += p[k];
    nmask[i] = s;
}

// gate dots: gf = c . Wg[:D], gs = c . Wg[D:2D]   (one block per c-row)
__global__ void gate_kernel(const float* __restrict__ c, const float* __restrict__ Wg,
                            float* __restrict__ gf, float* __restrict__ gs)
{
    __shared__ float sf[256], ss[256];
    int bc = blockIdx.x;
    const float* cr = c + (size_t)bc * Dd;
    float af = 0.f, as = 0.f;
    for (int k = threadIdx.x; k < Dd; k += 256) {
        float v = cr[k];
        af += v * Wg[k];
        as += v * Wg[Dd + k];
    }
    sf[threadIdx.x] = af; ss[threadIdx.x] = as;
    __syncthreads();
    for (int o = 128; o; o >>= 1) {
        if (threadIdx.x < o) { sf[threadIdx.x] += sf[threadIdx.x+o]; ss[threadIdx.x] += ss[threadIdx.x+o]; }
        __syncthreads();
    }
    if (threadIdx.x == 0) { gf[bc] = sf[0]; gs[bc] = ss[0]; }
}

// ---------------------------------------------------------------------------
// 2) conversions to f16 + pair row-map
// ---------------------------------------------------------------------------
__global__ void cvt_rows_kernel(const float* __restrict__ src, _Float16* __restrict__ dst, int n)
{
    for (int i = blockIdx.x * blockDim.x + threadIdx.x; i < n; i += gridDim.x * blockDim.x)
        dst[i] = (_Float16)src[i];
}

// Wfs[256][1536]: n<128 -> W1[n][k] ; else W1[n-128][D+k].  Wp[128][1536]: W1[n][2D+k]. W2h[128][128].
__global__ void cvt_w_kernel(const float* __restrict__ W1, const float* __restrict__ W2,
                             _Float16* __restrict__ Wfs, _Float16* __restrict__ Wp,
                             _Float16* __restrict__ W2h)
{
    const int NW1 = 256*Dd, NW2 = 128*Dd, NW3 = Hh*Hh;
    for (int i = blockIdx.x * blockDim.x + threadIdx.x; i < NW1 + NW2 + NW3;
         i += gridDim.x * blockDim.x) {
        if (i < NW1) {
            int n = i / Dd, k = i % Dd;
            Wfs[i] = (_Float16)(n < Hh ? W1[(size_t)n*3*Dd + k] : W1[(size_t)(n-Hh)*3*Dd + Dd + k]);
        } else if (i < NW1 + NW2) {
            int j = i - NW1, n = j / Dd, k = j % Dd;
            Wp[j] = (_Float16)W1[(size_t)n*3*Dd + 2*Dd + k];
        } else {
            int j = i - NW1 - NW2;
            W2h[j] = (_Float16)W2[j];
        }
    }
}

__global__ void rowmap_kernel(const int* __restrict__ fi, const int* __restrict__ si,
                              int2* __restrict__ rmap)
{
    int r = blockIdx.x * blockDim.x + threadIdx.x;
    if (r >= MBIG) return;
    if (r < NCROWS) {
        int b = r / NP, p = r % NP;
        rmap[r] = make_int2(b*NC + fi[p], b*NC + si[p]);
    } else {
        int q = r - NCROWS;
        int b = q / (NP*25), rem = q % (NP*25);
        int p = rem / 25, t = rem % 25;
        int a = t / 5, n2 = t % 5;
        rmap[r] = make_int2(Bz*NC + (b*NC + fi[p])*NNB + a,
                            Bz*NC + (b*NC + si[p])*NNB + n2);
    }
}

// ---------------------------------------------------------------------------
// 3) strip-mined WMMA GEMM, 4-fragment ping-pong pipeline:
//    C[M][N] = A[M][K]_f16 @ B.T with Bw stored [N][K] (K compile-time).
//    One wave computes a 16(M) x 128(N) strip with 8 f32 accumulators; the 8
//    n-tiles are processed as two groups of 4 (bA/bB ping-pong) so every load
//    group is covered by 4 in-flight WMMAs.  Live set ~150 VGPRs -> fits in
//    the direct 256-VGPR window (no s_set_vgpr_msb, no spills).
//    M mult of 16, N mult of 128, K mult of 64.
// ---------------------------------------------------------------------------
template<int K>
__global__ __launch_bounds__(128, 1)
void wmma_strip_gemm_kernel(const _Float16* __restrict__ A,
                            const _Float16* __restrict__ Bw,
                            float* __restrict__ C, int M, int N)
{
    int wave = (blockIdx.x * blockDim.x + threadIdx.x) >> 5;
    int lane = threadIdx.x & 31;
    int nstrips = N >> 7;
    int tm = wave / nstrips, strip = wave % nstrips;
    if (tm * 16 >= M) return;
    const _Float16* arow = A + (size_t)(tm*16 + (lane & 15)) * K;
    const _Float16* bk   = Bw + ((size_t)strip*128 + (lane & 15)) * K + ((lane >> 4) * 8);
    int ka = (lane >> 4) * 8;     // A: lanes 16-31 carry K+8..15 / K+24..31

    v8f acc[8] = {};
    V16 a0, a1, bA[4], bB[4];
    load_a(a0, arow, 0, ka);
    load_b4<K>(bA, bk, 0, 0);
    int k0 = 0;
    for (; k0 + 64 < K; k0 += 64) {
        load_b4<K>(bB, bk, k0, 4);
        load_a(a1, arow, k0 + 32, ka);
        wmma4(acc,     a0, bA);
        load_b4<K>(bA, bk, k0 + 32, 0);
        wmma4(acc + 4, a0, bB);
        load_b4<K>(bB, bk, k0 + 32, 4);
        load_a(a0, arow, k0 + 64, ka);
        wmma4(acc,     a1, bA);
        load_b4<K>(bA, bk, k0 + 64, 0);
        wmma4(acc + 4, a1, bB);
    }
    load_b4<K>(bB, bk, k0, 4);
    load_a(a1, arow, k0 + 32, ka);
    wmma4(acc,     a0, bA);
    load_b4<K>(bA, bk, k0 + 32, 0);
    wmma4(acc + 4, a0, bB);
    load_b4<K>(bB, bk, k0 + 32, 4);
    wmma4(acc,     a1, bA);
    wmma4(acc + 4, a1, bB);

    int rb = tm*16 + 8*(lane >> 4);
    #pragma unroll
    for (int n = 0; n < 8; ++n) {
        int col = strip*128 + n*16 + (lane & 15);
        #pragma unroll
        for (int v = 0; v < 8; ++v)
            C[(size_t)(rb + v) * N + col] = acc[n][v];
    }
}

// product-GEMM strip: A row r = rows[map[r].x] * rows[map[r].y] (elementwise),
// N = 128 (one strip). Same ping-pong pipeline; the pair product is built
// once per K-chunk (v_pk_mul_f16) and reused by 8 WMMAs.
template<int K>
__global__ __launch_bounds__(128, 1)
void wmma_pair_gemm_kernel(const _Float16* __restrict__ rows,
                           const int2* __restrict__ rmap,
                           const _Float16* __restrict__ Wp,  // [128][K]
                           float* __restrict__ P, int M)
{
    int wave = (blockIdx.x * blockDim.x + threadIdx.x) >> 5;
    int lane = threadIdx.x & 31;
    if (wave * 16 >= M) return;
    int2 fs = rmap[wave*16 + (lane & 15)];
    const _Float16* fr = rows + (size_t)fs.x * K;
    const _Float16* sr = rows + (size_t)fs.y * K;
    const _Float16* bk = Wp + (size_t)(lane & 15) * K + ((lane >> 4) * 8);
    int ka = (lane >> 4) * 8;

    v8f acc[8] = {};
    V16 a0, a1, bA[4], bB[4];
    load_pa(a0, fr, sr, 0, ka);
    load_b4<K>(bA, bk, 0, 0);
    int k0 = 0;
    for (; k0 + 64 < K; k0 += 64) {
        load_b4<K>(bB, bk, k0, 4);
        load_pa(a1, fr, sr, k0 + 32, ka);
        wmma4(acc,     a0, bA);
        load_b4<K>(bA, bk, k0 + 32, 0);
        wmma4(acc + 4, a0, bB);
        load_b4<K>(bB, bk, k0 + 32, 4);
        load_pa(a0, fr, sr, k0 + 64, ka);
        wmma4(acc,     a1, bA);
        load_b4<K>(bA, bk, k0 + 64, 0);
        wmma4(acc + 4, a1, bB);
    }
    load_b4<K>(bB, bk, k0, 4);
    load_pa(a1, fr, sr, k0 + 32, ka);
    wmma4(acc,     a0, bA);
    load_b4<K>(bA, bk, k0 + 32, 0);
    wmma4(acc + 4, a0, bB);
    load_b4<K>(bB, bk, k0 + 32, 4);
    wmma4(acc,     a1, bA);
    wmma4(acc + 4, a1, bB);

    int rb = wave*16 + 8*(lane >> 4);
    #pragma unroll
    for (int n = 0; n < 8; ++n) {
        int col = n*16 + (lane & 15);
        #pragma unroll
        for (int v = 0; v < 8; ++v)
            P[(size_t)(rb + v) * Hh + col] = acc[n][v];
    }
}

// ---------------------------------------------------------------------------
// 4) layer-1 combine: h1 = relu(P + projF(first) + projS(second) + b1) -> f16
// ---------------------------------------------------------------------------
__global__ void combine_kernel(const float* __restrict__ P,
                               const float* __restrict__ Cp,    // [NROWS][256]
                               const int2* __restrict__ rmap,
                               const float* __restrict__ b1,
                               _Float16* __restrict__ h1)
{
    int idx = blockIdx.x * blockDim.x + threadIdx.x;
    if (idx >= MBIG * Hh) return;
    int r = idx >> 7, n = idx & 127;
    int2 fs = rmap[r];
    float v = P[idx] + Cp[(size_t)fs.x * 256 + n] + Cp[(size_t)fs.y * 256 + Hh + n] + b1[n];
    h1[idx] = (_Float16)fmaxf(v, 0.f);
}

// score = relu(h2 + b2) . W3 + b3   (one wave per row)
__global__ void score_kernel(const float* __restrict__ P2, const float* __restrict__ b2,
                             const float* __restrict__ W3, const float* __restrict__ b3,
                             float* __restrict__ scores)
{
    int row  = (blockIdx.x * blockDim.x + threadIdx.x) >> 5;
    int lane = threadIdx.x & 31;
    if (row >= MBIG) return;
    float s = 0.f;
    for (int n = lane; n < Hh; n += 32)
        s += fmaxf(P2[(size_t)row * Hh + n] + b2[n], 0.f) * W3[n];
    #pragma unroll
    for (int o = 16; o; o >>= 1) s += __shfl_xor(s, o, 32);
    if (lane == 0) scores[row] = s + b3[0];
}

// ---------------------------------------------------------------------------
// 5) masked max/mean over 5x5 + gate blend
// ---------------------------------------------------------------------------
__global__ void final_kernel(const float* __restrict__ scores,
                             const float* __restrict__ nmask,
                             const int* __restrict__ fi, const int* __restrict__ si,
                             const float* __restrict__ gf, const float* __restrict__ gs,
                             const float* __restrict__ bg, float* __restrict__ out)
{
    int idx = blockIdx.x * blockDim.x + threadIdx.x;
    if (idx >= NCROWS) return;
    int b = idx / NP, p = idx % NP;
    int f = fi[p], s2 = si[p];
    float sc = scores[idx];                            // center-pair score
    size_t base = (size_t)NCROWS + (size_t)idx * 25;
    float msum = 0.f, rcnt = 0.f;
    for (int a = 0; a < NNB; ++a) {
        float mf = nmask[(b*NC + f)*NNB + a];
        float best = -3.402823466e38f;
        int has = 0;
        for (int n2 = 0; n2 < NNB; ++n2) {
            float am = mf * nmask[(b*NC + s2)*NNB + n2];
            if (am > 0.f) {
                float v = scores[base + a*5 + n2] * am;
                best = fmaxf(best, v);
                has = 1;
            }
        }
        if (has) { msum += best; rcnt += 1.f; }
    }
    float sn = msum / fmaxf(rcnt, 1e-13f);
    float z = gf[b*NC + f] + gs[b*NC + s2] + bg[0];
    float gate = 1.f / (1.f + __expf(-z));
    out[idx] = gate * sc + (1.f - gate) * sn;
}

// ---------------------------------------------------------------------------
extern "C" void kernel_launch(void* const* d_in, const int* in_sizes, int n_in,
                              void* d_out, int out_size, void* d_ws, size_t ws_size,
                              hipStream_t stream)
{
    const float* x    = (const float*)d_in[0];
    const float* cmap = (const float*)d_in[1];
    const float* nmap = (const float*)d_in[2];
    const int*   fi   = (const int*)  d_in[3];
    const int*   si   = (const int*)  d_in[4];
    const float* W1   = (const float*)d_in[5];
    const float* b1   = (const float*)d_in[6];
    const float* W2   = (const float*)d_in[7];
    const float* b2   = (const float*)d_in[8];
    const float* W3   = (const float*)d_in[9];
    const float* b3   = (const float*)d_in[10];
    const float* Wg   = (const float*)d_in[11];
    const float* bg   = (const float*)d_in[12];
    float* out = (float*)d_out;

    char* ws = (char*)d_ws;
    size_t o = 0;
    float*    c_f   = (float*)(ws + o);    o += (size_t)Bz*NC*Dd*4;          // 1.18 MB
    float*    nb_f  = (float*)(ws + o);    o += (size_t)Bz*NC*NNB*Dd*4;      // 5.90 MB (contiguous after c_f)
    _Float16* rowsH = (_Float16*)(ws + o); o += (size_t)NROWS*Dd*2;          // 3.54 MB
    _Float16* WfsH  = (_Float16*)(ws + o); o += (size_t)256*Dd*2;            // 0.79 MB
    _Float16* WpH   = (_Float16*)(ws + o); o += (size_t)Hh*Dd*2;             // 0.39 MB
    _Float16* W2H   = (_Float16*)(ws + o); o += (size_t)Hh*Hh*2;             // 32 KB
    float*    Cp    = (float*)(ws + o);    o += (size_t)NROWS*256*4;         // 1.18 MB
    int2*     rmap  = (int2*)(ws + o);     o += (size_t)MBIG*8;              // 0.46 MB
    float*    P1    = (float*)(ws + o);    o += (size_t)MBIG*Hh*4;           // 29.4 MB (reused as P2)
    _Float16* h1    = (_Float16*)(ws + o); o += (size_t)MBIG*Hh*2;           // 14.7 MB
    float*    scr   = (float*)(ws + o);    o += (size_t)MBIG*4;              // 0.23 MB
    float*    nmask = (float*)(ws + o);    o += (size_t)Bz*NC*NNB*4;
    float*    gfv   = (float*)(ws + o);    o += (size_t)Bz*NC*4;
    float*    gsv   = (float*)(ws + o);    o += (size_t)Bz*NC*4;
    (void)ws_size; (void)in_sizes; (void)n_in; (void)out_size;

    // stage 1: aggregation, masks, gate dots
    agg_kernel<<<Bz*NC, 256, 0, stream>>>(x, cmap, nmap, c_f, nb_f);
    mask_kernel<<<(Bz*NC*NNB + 255)/256, 256, 0, stream>>>(nmap, nmask);
    gate_kernel<<<Bz*NC, 256, 0, stream>>>(c_f, Wg, gfv, gsv);

    // stage 2: conversions + row map
    cvt_rows_kernel<<<4096, 256, 0, stream>>>(c_f, rowsH, NROWS*Dd);  // c_f/nb_f contiguous
    cvt_w_kernel<<<2048, 256, 0, stream>>>(W1, W2, WfsH, WpH, W2H);
    rowmap_kernel<<<(MBIG + 255)/256, 256, 0, stream>>>(fi, si, rmap);

    // stage 3: WMMA GEMMs (one wave = 16x128 strip, 8 accumulators, ping-pong pipeline)
    // projections: Cp[1152][256] = rowsH[1152][1536] @ [Wf;Ws].T  -> 144 waves
    wmma_strip_gemm_kernel<Dd><<<(NROWS/16)*(256/128)/4, 128, 0, stream>>>(
        rowsH, WfsH, Cp, NROWS, 256);
    // product term: P1[57408][128] = (rows[f]*rows[s]) @ Wp.T -> 3588 waves
    wmma_pair_gemm_kernel<Dd><<<(MBIG/16)/4, 128, 0, stream>>>(rowsH, rmap, WpH, P1, MBIG);
    // layer-1 combine -> h1 (f16)
    combine_kernel<<<(MBIG*Hh + 255)/256, 256, 0, stream>>>(P1, Cp, rmap, b1, h1);
    // layer-2: P2 = h1 @ W2.T  (reuse P1 storage) -> 3588 waves
    wmma_strip_gemm_kernel<Hh><<<(MBIG/16)/4, 128, 0, stream>>>(h1, W2H, P1, MBIG, Hh);
    // layer-3 + scores
    score_kernel<<<(MBIG*32 + 255)/256, 256, 0, stream>>>(P1, b2, W3, b3, scr);

    // stage 4: masked max/mean + gate blend
    final_kernel<<<(NCROWS + 255)/256, 256, 0, stream>>>(scr, nmask, fi, si, gfv, gsv, bg, out);
}
